// SelfAttentionHead_71494025610069
// MI455X (gfx1250) — matrix operand order repaired and checked by reading, not compiled, not run on hardware
//
#include <hip/hip_runtime.h>
#include <hip/hip_bf16.h>

typedef _Float16 v16h __attribute__((ext_vector_type(16)));
typedef _Float16 h8   __attribute__((ext_vector_type(8)));
typedef float    v8f  __attribute__((ext_vector_type(8)));
typedef int      v4i  __attribute__((ext_vector_type(4)));

#define WMMA_F16(a, b, c) \
  __builtin_amdgcn_wmma_f32_16x16x32_f16(false, (a), false, (b), (short)0, (c), false, false)

#ifndef __has_builtin
#define __has_builtin(x) 0
#endif
#if __has_builtin(__builtin_amdgcn_global_load_async_to_lds_b128)
#define HAVE_ASYNC_LDS 1
typedef __attribute__((address_space(1))) v4i gv4i;   // global int4
typedef __attribute__((address_space(3))) v4i lv4i;   // LDS int4
#else
#define HAVE_ASYNC_LDS 0
#endif

static constexpr int  BATCH = 4;
static constexpr int  T     = 4096;
static constexpr int  C     = 1024;
static constexpr int  HS    = 64;
static constexpr float SCALE = 0.03125f;   // C^-0.5 = 1/sqrt(1024)
static constexpr float NEGINF = -1e30f;

// Build a 16-element f16 fragment from two 16-byte chunks (two b128 loads).
__device__ __forceinline__ v16h frag2(const _Float16* p0, const _Float16* p1) {
  union { v16h v; h8 h[2]; } u;
  u.h[0] = *(const h8*)p0;
  u.h[1] = *(const h8*)p1;
  return u.v;
}

// 16-byte global -> LDS copy: async (ASYNCcnt) when available, sync fallback.
__device__ __forceinline__ void cp16_g2l(const _Float16* g, _Float16* l) {
#if HAVE_ASYNC_LDS
  __builtin_amdgcn_global_load_async_to_lds_b128((gv4i*)(g), (lv4i*)(l), 0, 0);
#else
  *(h8*)l = *(const h8*)g;
#endif
}

__device__ __forceinline__ void wait_async_lds() {
#if HAVE_ASYNC_LDS
#if __has_builtin(__builtin_amdgcn_s_wait_asynccnt)
  __builtin_amdgcn_s_wait_asynccnt(0);
#else
  asm volatile("s_wait_asynccnt 0" ::: "memory");
#endif
#endif
}

// ---------------------------------------------------------------------------
// Kernel 1: fused QKV projection.  [B*T, C] x [C, 64] -> k,q,v (f16).
// Block = 256 thr (8 waves).  Block tile: 32 rows x 64 cols x {k,q,v}.
// Wave w: row-tile = w&1, col-tile = w>>1; A fragment shared by 3 WMMAs.
// q is pre-scaled by C^-0.5; v is stored transposed [B][HS][T].
// ---------------------------------------------------------------------------
__global__ __launch_bounds__(256, 2)
void qkv_proj_kernel(const float* __restrict__ x,
                     const float* __restrict__ Wk, const float* __restrict__ bk,
                     const float* __restrict__ Wq, const float* __restrict__ bq,
                     const float* __restrict__ Wv, const float* __restrict__ bv,
                     _Float16* __restrict__ kbuf,
                     _Float16* __restrict__ qbuf,
                     _Float16* __restrict__ vT)
{
  __shared__ _Float16 xs[32][40];        // 32 rows x 32 K (padded)
  __shared__ _Float16 wl[3][64][40];     // per matrix: 64 cols x 32 K (transposed)

  const int tid  = threadIdx.x;
  const int lane = tid & 31;
  const int w    = tid >> 5;
  const int mt   = w & 1;                // 16-row tile within 32
  const int nt   = w >> 1;               // 16-col tile within 64
  const int l15  = lane & 15;
  const int hi   = lane >> 4;
  const int row0 = blockIdx.x * 32;

  const float* Ws[3] = {Wk, Wq, Wv};

  v8f acc[3] = {};

  for (int k0 = 0; k0 < C; k0 += 32) {
    __syncthreads();
    // stage x tile (fp32 -> f16), coalesced along K
    #pragma unroll
    for (int i = 0; i < 4; ++i) {
      int f = tid + i * 256;
      int r = f >> 5, kk = f & 31;
      xs[r][kk] = (_Float16)x[(size_t)(row0 + r) * C + k0 + kk];
    }
    // stage W tiles, transposed in LDS so B-fragments are contiguous in K
    #pragma unroll
    for (int m = 0; m < 3; ++m) {
      #pragma unroll
      for (int i = 0; i < 8; ++i) {
        int f = tid + i * 256;
        int n = f & 63, cr = f >> 6;
        wl[m][n][cr] = (_Float16)Ws[m][(size_t)(k0 + cr) * HS + n];
      }
    }
    __syncthreads();

    // A: 16x32 f16 fragment (lane<16: K 0-7 & 16-23; lane>=16: K 8-15 & 24-31)
    v16h a = frag2(&xs[mt * 16 + l15][hi * 8],
                   &xs[mt * 16 + l15][16 + hi * 8]);
    #pragma unroll
    for (int m = 0; m < 3; ++m) {
      // B: 32x16, lane<16: K 0-7 & 16-23; lane>=16: K 8-15 & 24-31
      v16h bf = frag2(&wl[m][nt * 16 + l15][hi * 8],
                      &wl[m][nt * 16 + l15][16 + hi * 8]);
      acc[m] = WMMA_F16(a, bf, acc[m]);
    }
  }

  const int col = nt * 16 + l15;
  const float bias0 = bk[col], bias1 = bq[col], bias2 = bv[col];
  #pragma unroll
  for (int r = 0; r < 8; ++r) {
    int rowg = row0 + mt * 16 + hi * 8 + r;       // global row in [0, B*T)
    float kv = acc[0][r] + bias0;
    float qv = (acc[1][r] + bias1) * SCALE;       // fold softmax scale into q
    float vv = acc[2][r] + bias2;
    kbuf[(size_t)rowg * HS + col] = (_Float16)kv;
    qbuf[(size_t)rowg * HS + col] = (_Float16)qv;
    int b = rowg >> 12, t = rowg & (T - 1);
    vT[(size_t)(b * HS + col) * T + t] = (_Float16)vv;   // transposed store
  }
}

// ---------------------------------------------------------------------------
// Kernel 2: causal flash attention.  Block = 128 thr (4 waves), grid (T/64, B).
// K/V tiles are staged once per block into double-buffered LDS with async
// global->LDS copies (ASYNCcnt), overlapped with WMMA compute on prev tile.
// ---------------------------------------------------------------------------
__global__ __launch_bounds__(128, 4)
void flash_attn_kernel(const _Float16* __restrict__ qbuf,
                       const _Float16* __restrict__ kbuf,
                       const _Float16* __restrict__ vT,
                       float* __restrict__ out)
{
  __shared__ _Float16 kls[2][64][72];    // K tile: [key][hs], double buffered
  __shared__ _Float16 vls[2][64][72];    // V tile: [hs][key], double buffered
  __shared__ _Float16 pls[4][16][72];    // per-wave P tile (16 rows x 64)

  const int tid  = threadIdx.x;
  const int lane = tid & 31;
  const int w    = tid >> 5;
  const int l15  = lane & 15;
  const int hi   = lane >> 4;
  const int qt   = blockIdx.x;           // query tile index (64 rows)
  const int b    = blockIdx.y;
  const int qrow0 = qt * 64 + w * 16;    // wave's first query row (in batch)
  const size_t rbase = (size_t)b * T;
  const _Float16* kg0 = kbuf + rbase * HS;           // K rows for this batch
  const _Float16* vg0 = vT + (size_t)b * HS * T;     // V^T rows for this batch

  // Q fragments (K = 0..31 and 32..63), loaded once
  const _Float16* qp = qbuf + (rbase + qrow0 + l15) * HS;
  v16h aQ[2];
  aQ[0] = frag2(qp + hi * 8,      qp + 16 + hi * 8);
  aQ[1] = frag2(qp + 32 + hi * 8, qp + 48 + hi * 8);

  v8f   O[4] = {};
  float mrow[8], lrow[8];
  #pragma unroll
  for (int r = 0; r < 8; ++r) { mrow[r] = NEGINF; lrow[r] = 0.f; }

  // prologue: stage tile j=0 into buffer 0 (4 K + 4 V chunks of 16B per thread)
  #pragma unroll
  for (int i = 0; i < 4; ++i) {
    int c = tid + i * 128;               // 0..511
    int row = c >> 3, off = (c & 7) * 8;
    cp16_g2l(kg0 + (size_t)row * HS + off, &kls[0][row][off]);
  }
  #pragma unroll
  for (int i = 0; i < 4; ++i) {
    int c = tid + i * 128;
    int row = c >> 3, off = (c & 7) * 8;
    cp16_g2l(vg0 + (size_t)row * T + off, &vls[0][row][off]);
  }

  for (int j = 0; j <= qt; ++j) {
    const int jbase = j * 64;
    const int buf = j & 1;

    wait_async_lds();                    // own async writes for tile j landed
    __syncthreads();                     // all waves' writes landed

    // overlap: stream tile j+1 into the other buffer while computing tile j
    if (j < qt) {
      const int nb = jbase + 64;
      #pragma unroll
      for (int i = 0; i < 4; ++i) {
        int c = tid + i * 128;
        int row = c >> 3, off = (c & 7) * 8;
        cp16_g2l(kg0 + (size_t)(nb + row) * HS + off, &kls[buf ^ 1][row][off]);
      }
      #pragma unroll
      for (int i = 0; i < 4; ++i) {
        int c = tid + i * 128;
        int row = c >> 3, off = (c & 7) * 8;
        cp16_g2l(vg0 + (size_t)row * T + nb + off, &vls[buf ^ 1][row][off]);
      }
    }

    // --- S = Q * K^T (scale pre-folded into q), B-fragments from LDS ---
    v8f s[4] = {};
    #pragma unroll
    for (int nt = 0; nt < 4; ++nt) {
      const _Float16* kp = &kls[buf][nt * 16 + l15][0];
      #pragma unroll
      for (int ks = 0; ks < 2; ++ks) {
        v16h bK = frag2(kp + ks * 32 + hi * 8, kp + ks * 32 + 16 + hi * 8);
        s[nt] = WMMA_F16(aQ[ks], bK, s[nt]);
      }
    }

    // --- causal mask on the diagonal tile ---
    if (j == qt) {
      #pragma unroll
      for (int nt = 0; nt < 4; ++nt) {
        int colg = jbase + nt * 16 + l15;
        #pragma unroll
        for (int r = 0; r < 8; ++r) {
          int rowg = qrow0 + hi * 8 + r;
          if (colg > rowg) s[nt][r] = NEGINF;
        }
      }
    }

    // --- online softmax (row = VGPR index, cols striped over 16-lane half) ---
    #pragma unroll
    for (int r = 0; r < 8; ++r) {
      float tm = fmaxf(fmaxf(s[0][r], s[1][r]), fmaxf(s[2][r], s[3][r]));
      #pragma unroll
      for (int m = 1; m < 16; m <<= 1) tm = fmaxf(tm, __shfl_xor(tm, m, 32));
      float mn    = fmaxf(mrow[r], tm);
      float alpha = __expf(mrow[r] - mn);
      float rs = 0.f;
      #pragma unroll
      for (int nt = 0; nt < 4; ++nt) {
        float p = __expf(s[nt][r] - mn);
        s[nt][r] = p;
        rs += p;
      }
      #pragma unroll
      for (int m = 1; m < 16; m <<= 1) rs += __shfl_xor(rs, m, 32);
      lrow[r] = lrow[r] * alpha + rs;
      mrow[r] = mn;
      #pragma unroll
      for (int nt = 0; nt < 4; ++nt) O[nt][r] *= alpha;
    }

    // --- stage P to LDS (C-layout -> A-layout relayout), same-wave only ---
    #pragma unroll
    for (int nt = 0; nt < 4; ++nt)
      #pragma unroll
      for (int r = 0; r < 8; ++r)
        pls[w][hi * 8 + r][nt * 16 + l15] = (_Float16)s[nt][r];

    const _Float16* pp = &pls[w][l15][0];   // LDS in-order within wave
    v16h aP[2];
    aP[0] = frag2(pp + hi * 8,      pp + 16 + hi * 8);
    aP[1] = frag2(pp + 32 + hi * 8, pp + 48 + hi * 8);

    // --- O += P * V, B-fragments from LDS (V^T: contiguous along key) ---
    #pragma unroll
    for (int nt = 0; nt < 4; ++nt) {
      const _Float16* vp = &vls[buf][nt * 16 + l15][0];
      #pragma unroll
      for (int ks = 0; ks < 2; ++ks) {
        v16h bV = frag2(vp + ks * 32 + hi * 8, vp + ks * 32 + 16 + hi * 8);
        O[nt] = WMMA_F16(aP[ks], bV, O[nt]);
      }
    }
  }

  // --- epilogue: out = O / l ---
  #pragma unroll
  for (int r = 0; r < 8; ++r) {
    float inv = 1.f / lrow[r];
    int t = qrow0 + hi * 8 + r;
    #pragma unroll
    for (int nt = 0; nt < 4; ++nt)
      out[(rbase + t) * HS + nt * 16 + l15] = O[nt][r] * inv;
  }
}

// ---------------------------------------------------------------------------
extern "C" void kernel_launch(void* const* d_in, const int* in_sizes, int n_in,
                              void* d_out, int out_size, void* d_ws, size_t ws_size,
                              hipStream_t stream) {
  const float* x  = (const float*)d_in[0];
  const float* Wk = (const float*)d_in[1];
  const float* bk = (const float*)d_in[2];
  const float* Wq = (const float*)d_in[3];
  const float* bq = (const float*)d_in[4];
  const float* Wv = (const float*)d_in[5];
  const float* bv = (const float*)d_in[6];
  float* out = (float*)d_out;

  const size_t N = (size_t)BATCH * T * HS;   // 1M elements per buffer
  _Float16* kbuf = (_Float16*)d_ws;
  _Float16* qbuf = kbuf + N;
  _Float16* vTb  = qbuf + N;                 // 6 MB total workspace

  qkv_proj_kernel<<<(BATCH * T) / 32, 256, 0, stream>>>(
      x, Wk, bk, Wq, bq, Wv, bv, kbuf, qbuf, vTb);

  flash_attn_kernel<<<dim3(T / 64, BATCH), 128, 0, stream>>>(
      qbuf, kbuf, vTb, out);
}
